// AllPoleDigitalFilter_14370960572882
// MI455X (gfx1250) — matrix operand order repaired and checked by reading, not compiled, not run on hardware
//
#include <hip/hip_runtime.h>

typedef __attribute__((ext_vector_type(16))) _Float16 v16h;
typedef __attribute__((ext_vector_type(8)))  float    v8f;

#define B_    32
#define N_    800
#define M_    24        // filter order
#define P_    80        // frame period
#define T_    (N_ * P_) // 64000 samples / batch
#define CHUNK 256       // samples per scan chunk
#define NSUB  8         // sub-chunks per chunk
#define SUBL  32        // steps per sub-chunk (8*32 = 256)
#define NC    (T_ / CHUNK) // 250 chunks / batch
#define ROWS  32        // padded state dim (24 -> 32 for WMMA tiles)
#define STR   33        // LDS row stride (bank-conflict free)
#define NFR   5         // frames staged per chunk (256/80 spans <=4 incl. lerp +1)

// ---------------------------------------------------------------------------
// Stage the <=5 coefficient frames a chunk touches into LDS via the CDNA5
// async global->LDS engine (GVS addressing: saddr pair + 32-bit voffset).
// cf[] holds NFR*25 floats: frame-major, channel 0 = gain, 1..24 = taps.
// ---------------------------------------------------------------------------
__device__ __forceinline__ void stage_frames(float* cf, const float* __restrict__ ab,
                                             int fbase, int tid, int nthreads) {
    unsigned long long gbase = (unsigned long long)(size_t)ab;
    for (int i = tid; i < NFR * 25; i += nthreads) {
        int f = fbase + i / 25;
        if (f > N_ - 1) f = N_ - 1;
        unsigned goff  = (unsigned)((f * 25 + i % 25) * 4);
        unsigned laddr = (unsigned)(size_t)&cf[i];
        asm volatile("global_load_async_to_lds_b32 %0, %1, %2"
                     :: "v"(laddr), "v"(goff), "s"(gbase) : "memory");
    }
    asm volatile("s_wait_asynccnt 0x0" ::: "memory");
}

// interpolated tap m at sample t, from staged LDS frames
__device__ __forceinline__ float tapL(const float* cf, int fbase, int t, int m) {
    int   f  = t / P_;
    float w  = (float)(t - f * P_) * (1.0f / (float)P_);
    int   f2 = (f + 1 < N_) ? f + 1 : N_ - 1;
    return cf[(f - fbase) * 25 + 1 + m] * (1.0f - w) + cf[(f2 - fbase) * 25 + 1 + m] * w;
}
// interpolated gain at sample t, from staged LDS frames
__device__ __forceinline__ float gainL(const float* cf, int fbase, int t) {
    int   f  = t / P_;
    float w  = (float)(t - f * P_) * (1.0f / (float)P_);
    int   f2 = (f + 1 < N_) ? f + 1 : N_ - 1;
    return cf[(f - fbase) * 25] * (1.0f - w) + cf[(f2 - fbase) * 25] * w;
}

// ---------------------------------------------------------------------------
// wave-level 32x32 f32 matmul C = A * B via v_wmma_f32_16x16x32_f16.
// A/B/C in LDS with row stride STR. Lane layouts per CDNA5 ISA.
// ---------------------------------------------------------------------------
__device__ __forceinline__ void wave_mm32(const float* __restrict__ A,
                                          const float* __restrict__ Bm,
                                          float* __restrict__ C, int lane) {
    const int l16 = lane & 15;
    const int hi  = lane >> 4; // 0 or 1
#pragma unroll
    for (int ti = 0; ti < 2; ++ti) {
#pragma unroll
        for (int tj = 0; tj < 2; ++tj) {
            v16h av, bv;
#pragma unroll
            for (int h = 0; h < 16; ++h) {
                int ka = (h & 7) + ((h & 8) ? 16 : 0) + hi * 8;
                av[h]  = (_Float16)A[(ti * 16 + l16) * STR + ka];
                int kb = (h & 7) + ((h & 8) ? 16 : 0) + hi * 8;
                bv[h]  = (_Float16)Bm[kb * STR + tj * 16 + l16];
            }
            v8f acc = {};
            acc = __builtin_amdgcn_wmma_f32_16x16x32_f16(
                false, av, false, bv, (short)0, acc, false, false);
#pragma unroll
            for (int r = 0; r < 8; ++r)
                C[(ti * 16 + r + hi * 8) * STR + tj * 16 + l16] = acc[r];
        }
    }
}

// wave-level out[i] = sum_k S[i][k]*u[k] + add[i]
__device__ __forceinline__ void wave_mv32(const float* __restrict__ S,
                                          const float* __restrict__ u,
                                          const float* __restrict__ add,
                                          float* __restrict__ out, int lane) {
    float acc = add[lane];
#pragma unroll 8
    for (int k = 0; k < ROWS; ++k) acc = fmaf(S[lane * STR + k], u[k], acc);
    out[lane] = acc;
}

// ---------------------------------------------------------------------------
// K2: per 256-sample chunk build the affine transfer (M, u): s_out = M s_in + u.
// 8 waves evolve 25 columns (24 homog + forced) over 32-step sub-chunks, then
// a 3-level WMMA tree combines: (S1,u1)o(S0,u0) = (S1*S0, S1*u0 + u1).
// M stored COLUMN-major to global so k_scan reads coalesced.
// ---------------------------------------------------------------------------
__global__ void __launch_bounds__(256) k_chunk(const float* __restrict__ x,
                                               const float* __restrict__ a,
                                               float* __restrict__ Mb,
                                               float* __restrict__ Ub) {
    __shared__ float cf[NFR * 25];
    __shared__ float Smat[NSUB][ROWS * STR];
    __shared__ float Pmat[4][ROWS * STR];
    __shared__ float Uvec[NSUB][ROWS];
    __shared__ float PU[4][ROWS];
    __shared__ float QU[2][ROWS];
    __shared__ float FU[ROWS];

    const int cid  = blockIdx.x; // b*NC + ck
    const int b    = cid / NC;
    const int ck   = cid - b * NC;
    const int tid  = threadIdx.x;
    const int wave = tid >> 5;
    const int lane = tid & 31;
    const float* ab = a + (size_t)b * N_ * 25;
    const float* xb = x + (size_t)b * T_;
    const int tC    = ck * CHUNK;
    const int fbase = tC / P_;

    stage_frames(cf, ab, fbase, tid, 256);
    __syncthreads();

    // ---- sub-chunk recurrence: lane j evolves column j of the transfer op ----
    float h[M_];
#pragma unroll
    for (int i = 0; i < M_; ++i) h[i] = (lane == i) ? 1.0f : 0.0f; // lanes>=24 zero

    const int t0 = tC + wave * SUBL;
    for (int s = 0; s < SUBL; ++s) {
        int   t   = t0 + s;
        float myc = (lane < M_) ? tapL(cf, fbase, t, lane) : 0.0f;
        float xv  = gainL(cf, fbase, t) * xb[t];
        float acc = 0.0f;
#pragma unroll
        for (int m = 0; m < M_; ++m) {
            float c = __shfl(myc, m, 32);
            acc = fmaf(c, h[m], acc);
        }
        float newv = ((lane == M_) ? xv : 0.0f) - acc; // lane 24 = forced column
#pragma unroll
        for (int i = M_ - 1; i > 0; --i) h[i] = h[i - 1];
        h[0] = newv;
    }
#pragma unroll
    for (int i = 0; i < M_; ++i)
        Smat[wave][i * STR + lane] = (lane < M_) ? h[i] : 0.0f;
#pragma unroll
    for (int i = M_; i < ROWS; ++i) Smat[wave][i * STR + lane] = 0.0f;
    if (lane == M_) {
#pragma unroll
        for (int i = 0; i < M_; ++i) Uvec[wave][i] = h[i];
    }
    if (lane == 0) {
#pragma unroll
        for (int i = M_; i < ROWS; ++i) Uvec[wave][i] = 0.0f;
    }
    __syncthreads();

    if (wave < 4) { // level 1
        wave_mm32(Smat[2 * wave + 1], Smat[2 * wave], Pmat[wave], lane);
        wave_mv32(Smat[2 * wave + 1], Uvec[2 * wave], Uvec[2 * wave + 1], PU[wave], lane);
    }
    __syncthreads();
    if (wave < 2) { // level 2 (reuse Smat slots)
        wave_mm32(Pmat[2 * wave + 1], Pmat[2 * wave], Smat[wave], lane);
        wave_mv32(Pmat[2 * wave + 1], PU[2 * wave], PU[2 * wave + 1], QU[wave], lane);
    }
    __syncthreads();
    if (wave == 0) { // level 3
        wave_mm32(Smat[1], Smat[0], Pmat[0], lane);
        wave_mv32(Smat[1], QU[0], QU[1], FU, lane);
    }
    __syncthreads();

    // coalesced COLUMN-major store: Md[k*32 + i] = M[i][k]
    float* Md = Mb + (size_t)cid * (ROWS * ROWS);
#pragma unroll
    for (int kk = 0; kk < (ROWS * ROWS) / 256; ++kk) {
        int idx = tid + kk * 256;
        int kcol = idx >> 5, irow = idx & 31;
        Md[idx] = Pmat[0][irow * STR + kcol]; // LDS banks (irow+kcol)%32: conflict-free
    }
    if (wave == 0) Ub[(size_t)cid * ROWS + lane] = FU[lane];
}

// ---------------------------------------------------------------------------
// K3: per-batch sequential affine scan over 250 chunk operators (column-major,
// coalesced), with prefetch of the next operator to hide latency.
// ---------------------------------------------------------------------------
__global__ void k_scan(const float* __restrict__ Mb, const float* __restrict__ Ub,
                       float* __restrict__ Hin) {
    const int b = blockIdx.x, lane = threadIdx.x;
    __shared__ float c[ROWS];
    c[lane] = 0.0f;
    __syncthreads();
    for (int ck = 0; ck < NC; ++ck) {
        size_t base = (size_t)b * NC + ck;
        Hin[base * ROWS + lane] = c[lane];
        const float* Mm = Mb + base * (ROWS * ROWS);
        if (ck + 1 < NC) // prefetch next 4KB operator (global_prefetch_b8)
            __builtin_prefetch(Mm + (ROWS * ROWS) + lane * ROWS, 0, 1);
        float acc = Ub[base * ROWS + lane];
#pragma unroll 8
        for (int k = 0; k < ROWS; ++k)
            acc = fmaf(Mm[k * ROWS + lane], c[k], acc); // coalesced per k
        __syncthreads();
        c[lane] = acc;
        __syncthreads();
    }
}

// ---------------------------------------------------------------------------
// K4: replay each chunk in f32 from its true entering state, scatter-to-future:
// lane l owns the accumulator of the future sample with residue l (mod 32).
// Per step: one broadcast shfl + one FMA on the dependence chain (vs a 5-level
// reduction), coefficient fetches (from LDS-staged frames) off the chain.
// ---------------------------------------------------------------------------
__global__ void k_replay(const float* __restrict__ x, const float* __restrict__ a,
                         const float* __restrict__ Hin, float* __restrict__ out) {
    __shared__ float cf[NFR * 25];
    const int cid  = blockIdx.x;
    const int b    = cid / NC;
    const int ck   = cid - b * NC;
    const int lane = threadIdx.x;
    const float* ab = a   + (size_t)b * N_ * 25;
    const float* xb = x   + (size_t)b * T_;
    float*       ob = out + (size_t)b * T_;
    const int t0    = ck * CHUNK;
    const int fbase = t0 / P_;

    stage_frames(cf, ab, fbase, lane, 32);
    __syncthreads();

    float hist = Hin[(size_t)cid * ROWS + lane]; // lane m: y[t0-1-m]
    if (lane >= M_) hist = 0.0f;

    // init accumulators with the entering-state (hist) contributions:
    // slot time tau = t0 + s0 receives -sum_{j>s0} a[tau][j-1] * hist[j-s0-1]
    float A  = 0.0f;
    int   s0 = (lane - t0) & 31;
#pragma unroll
    for (int j = 1; j <= M_; ++j) {
        float hv = __shfl(hist, (j - s0 - 1) & 31, 32);
        if (j > s0) A = fmaf(-tapL(cf, fbase, t0 + s0, j - 1), hv, A);
    }

    float outv = 0.0f;
    for (int s = 0; s < CHUNK; ++s) {
        int   t  = t0 + s;
        float xv = gainL(cf, fbase, t) * xb[t];
        float y  = xv + __shfl(A, t & 31, 32);      // broadcast finished slot
        if (lane == (t & 31)) { outv = y; A = 0.0f; } // slot now owns t+32
        int j = (lane - t) & 31;                    // my slot is time t+j
        if (j >= 1 && j <= M_ && (s + j) < CHUNK) { // cross-chunk handled by Hin
            float cc = tapL(cf, fbase, t + j, j - 1);
            A = fmaf(-cc, y, A);
        }
        if ((s & 31) == 31) ob[t0 + (s & ~31) + lane] = outv; // coalesced
    }
}

// ---------------------------------------------------------------------------
extern "C" void kernel_launch(void* const* d_in, const int* in_sizes, int n_in,
                              void* d_out, int out_size, void* d_ws, size_t ws_size,
                              hipStream_t stream) {
    (void)in_sizes; (void)n_in; (void)out_size; (void)ws_size;
    const float* x = (const float*)d_in[0]; // (B, T)
    const float* a = (const float*)d_in[1]; // (B, N, 25)
    float* out = (float*)d_out;             // (B, T)

    float* Mb  = (float*)d_ws;                         // B*NC*32*32 (col-major)
    float* Ub  = Mb + (size_t)B_ * NC * ROWS * ROWS;   // B*NC*32
    float* Hin = Ub + (size_t)B_ * NC * ROWS;          // B*NC*32

    k_chunk <<<B_ * NC, 256, 0, stream>>>(x, a, Mb, Ub);
    k_scan  <<<B_, 32, 0, stream>>>(Mb, Ub, Hin);
    k_replay<<<B_ * NC, 32, 0, stream>>>(x, a, Hin, out);
}
